// Convolution_11123965297047
// MI455X (gfx1250) — compile-verified, matrix operand order checked by reading, not run on hardware
//
#include <hip/hip_runtime.h>

// ---------------------------------------------------------------------------
// Problem constants
// ---------------------------------------------------------------------------
#define HW      262144      // 512*512 pixels per channel plane
#define WPIX    512
#define C_IN    256
#define TAPS    121         // 11*11
#define NPADT   128         // taps padded to 8 WMMA M-tiles
#define OD      502         // 512 - 10
#define OSZ     (OD * OD)   // 252004 outputs per batch image
#define QPB     128         // pixels (GEMM N) per workgroup
#define PW_BYTES 65536      // packed-weight region in workspace

typedef __attribute__((ext_vector_type(16))) __bf16 bf16x16;
typedef __attribute__((ext_vector_type(2)))  __bf16 bf16x2;
typedef __attribute__((ext_vector_type(8)))  float  f32x8;

// ---------------------------------------------------------------------------
// Kernel 1: pack weights into per-lane WMMA A-fragment layout.
//   A is M(tap) x K(channel).  Fragment (16-bit A 16x32):
//     lane L (lh = L>>4): element e -> K = (e<8 ? e : e+8) + 8*lh
//   pw[((kt*8+mt)*32 + L)*16 + e]  (bf16), 64 frags * 32 lanes * 32B = 64 KB
// ---------------------------------------------------------------------------
__global__ void prep_weights(const float* __restrict__ kern,
                             __bf16* __restrict__ pw) {
  int t    = blockIdx.x * blockDim.x + threadIdx.x;
  int frag = t >> 5;                 // 0..63 = kt*8 + mt
  if (frag >= 64) return;
  int L  = t & 31;
  int lh = L >> 4;
  int mt = frag & 7;
  int kt = frag >> 3;
  int m  = mt * 16 + (L & 15);       // tap index
#pragma unroll
  for (int e = 0; e < 16; ++e) {
    int K = ((e < 8) ? e : (e + 8)) + 8 * lh;
    int c = kt * 32 + K;             // channel
    float v = (m < TAPS) ? kern[c * TAPS + m] : 0.0f;
    pw[((size_t)frag * 32 + L) * 16 + e] = (__bf16)v;
  }
}

// ---------------------------------------------------------------------------
// Kernel 2: GEMM  partial[tap, pixel] = sum_c W[c,tap] * (x0+x1)[c,pixel]
//   M = 128 taps (8 tiles), N = 128 pixels per WG (16 per wave), K = 256 (8x32)
//   Software-pipelined: chunk kt+1 is in flight (global->VGPR) while chunk kt
//   runs its WMMAs; LDS double-buffered -> one barrier per k-step.
//   LDS holds the x chunk pre-arranged in B-fragment layout so each lane's
//   fragment read is 32 contiguous bytes.
// ---------------------------------------------------------------------------
#define SL 36                   // bf16 slot per lane (16 used + pad)
#define SW (32 * SL + 4)        // 1156 bf16 per wave block (578 dw == 2 mod 64)

__global__ void __launch_bounds__(256) conv_gemm(
    const float* __restrict__ x,        // [2][256][512*512] fp32
    const __bf16* __restrict__ pw,      // packed weights
    _Float16* __restrict__ partial) {   // [128][262144] f16, tap-major
  __shared__ __bf16 fragbuf[2][8 * SW];  // 2 x 18496 B

  const int  t    = threadIdx.x;
  const int  wave = t >> 5;
  const int  lane = t & 31;
  const long qbase = (long)blockIdx.x * QPB;

  // staging role: channel pair cp = t/16 (c = 2cp, 2cp+1), pixel octet qg = t%16
  const int cp  = t >> 4;                      // 0..15
  const int qg  = t & 15;                      // 0..15
  const int lhC = (cp >> 2) & 1;               // fragment lane-half (same for pair)
  const int e0  = (2 * (cp & 3)) | (cp & 8);   // even fragment element of pair
  const int wrBase =                           // bf16 index of pixel i=0 slot
      (qg >> 1) * SW + (8 * (qg & 1) + 16 * lhC) * SL + e0;

  f32x8 acc[8] = {};                           // 8 M-tiles of C

  // ---- pipelined global loads: r[0..3]=c0 (b0,b0,b1,b1), r[4..7]=c1 ----
  float4 r[8];
  {
    const size_t c0 = (size_t)(2 * cp);
    const float4* p00 = (const float4*)(x + c0 * HW + qbase + qg * 8);
    const float4* p01 = (const float4*)(x + (c0 + C_IN) * HW + qbase + qg * 8);
    const float4* p10 = (const float4*)(x + (c0 + 1) * HW + qbase + qg * 8);
    const float4* p11 = (const float4*)(x + (c0 + 1 + C_IN) * HW + qbase + qg * 8);
    r[0] = p00[0]; r[1] = p00[1]; r[2] = p01[0]; r[3] = p01[1];
    r[4] = p10[0]; r[5] = p10[1]; r[6] = p11[0]; r[7] = p11[1];
  }

  for (int kt = 0; kt < 8; ++kt) {
    // ---- commit chunk kt from registers into LDS fragment layout ----
    {
      float s0[8], s1[8];
#pragma unroll
      for (int i = 0; i < 4; ++i) {
        s0[i]     = ((const float*)&r[0])[i] + ((const float*)&r[2])[i];
        s0[i + 4] = ((const float*)&r[1])[i] + ((const float*)&r[3])[i];
        s1[i]     = ((const float*)&r[4])[i] + ((const float*)&r[6])[i];
        s1[i + 4] = ((const float*)&r[5])[i] + ((const float*)&r[7])[i];
      }
#pragma unroll
      for (int i = 0; i < 8; ++i) {
        bf16x2 p;
        p.x = (__bf16)s0[i];     // native packed f32->bf16 path
        p.y = (__bf16)s1[i];
        *(bf16x2*)&fragbuf[kt & 1][wrBase + i * SL] = p;
      }
    }

    // ---- kick off global loads for chunk kt+1 (consumed next iteration) ----
    if (kt < 7) {
      const size_t c0 = (size_t)((kt + 1) * 32 + 2 * cp);
      const float4* p00 = (const float4*)(x + c0 * HW + qbase + qg * 8);
      const float4* p01 = (const float4*)(x + (c0 + C_IN) * HW + qbase + qg * 8);
      const float4* p10 = (const float4*)(x + (c0 + 1) * HW + qbase + qg * 8);
      const float4* p11 = (const float4*)(x + (c0 + 1 + C_IN) * HW + qbase + qg * 8);
      r[0] = p00[0]; r[1] = p00[1]; r[2] = p01[0]; r[3] = p01[1];
      r[4] = p10[0]; r[5] = p10[1]; r[6] = p11[0]; r[7] = p11[1];
    }

    __syncthreads();   // stores of chunk kt visible; buffers alternate

    // ---- compute: preload 8 A-frags in one clause, B from LDS, 8 WMMAs ----
    {
      union { uint4 q[2]; bf16x16 v; } A[8];
      const uint4* pA = (const uint4*)pw + (size_t)kt * 512 + lane * 2;
#pragma unroll
      for (int mt = 0; mt < 8; ++mt) {
        A[mt].q[0] = pA[mt * 64];
        A[mt].q[1] = pA[mt * 64 + 1];
      }

      union { unsigned long long u[4]; bf16x16 v; } B;
      const unsigned long long* pb =
          (const unsigned long long*)&fragbuf[kt & 1][wave * SW + lane * SL];
      B.u[0] = pb[0]; B.u[1] = pb[1]; B.u[2] = pb[2]; B.u[3] = pb[3];

#pragma unroll
      for (int mt = 0; mt < 8; ++mt) {
        acc[mt] = __builtin_amdgcn_wmma_f32_16x16x32_bf16(
            false, A[mt].v, false, B.v, (short)0, acc[mt], false, false);
      }
    }
  }

  // ---- store C: lane L -> N(pixel) = L%16, M(tap) = r + 8*(L>>4) ----
  {
    const int  n  = lane & 15;
    const int  lh = lane >> 4;
    const long q  = qbase + wave * 16 + n;
#pragma unroll
    for (int mt = 0; mt < 8; ++mt) {
#pragma unroll
      for (int rr = 0; rr < 8; ++rr) {
        int m = mt * 16 + rr + 8 * lh;
        partial[(size_t)m * HW + q] = (_Float16)acc[mt][rr];
      }
    }
  }
}

// ---------------------------------------------------------------------------
// Kernel 3: gather/col2im.  out[i,j] = bias + sum_{u,v} partial[u*11+v,
//           (i+u)*512 + (j+v)].  Tap-major partial => all 121 loads coalesced.
//           Same scalar map broadcast to both batch slots.
// ---------------------------------------------------------------------------
__global__ void __launch_bounds__(256) conv_gather(
    const _Float16* __restrict__ partial,
    const float* __restrict__ bias,
    float* __restrict__ out) {
  int tid = blockIdx.x * blockDim.x + threadIdx.x;
  if (tid >= OSZ) return;
  int i = tid / OD;
  int j = tid - i * OD;
  float accv = 0.0f;
#pragma unroll
  for (int u = 0; u < 11; ++u) {
    const _Float16* pr =
        partial + (size_t)(u * 11) * HW + (size_t)(i + u) * WPIX + j;
#pragma unroll
    for (int v = 0; v < 11; ++v) {
      accv += (float)pr[(size_t)v * HW + v];
    }
  }
  float rr = accv + bias[0];
  out[tid]       = rr;
  out[OSZ + tid] = rr;
}

// ---------------------------------------------------------------------------
// Fallback (workspace too small): direct conv, one thread per output.
// ---------------------------------------------------------------------------
__global__ void conv_naive(const float* __restrict__ x,
                           const float* __restrict__ kern,
                           const float* __restrict__ bias,
                           float* __restrict__ out) {
  int tid = blockIdx.x * blockDim.x + threadIdx.x;
  if (tid >= OSZ) return;
  int i = tid / OD;
  int j = tid - i * OD;
  float acc = 0.0f;
  for (int c = 0; c < C_IN; ++c) {
    const float* x0 = x + (size_t)c * HW;
    const float* x1 = x0 + (size_t)C_IN * HW;
    const float* w  = kern + c * TAPS;
    for (int u = 0; u < 11; ++u) {
      size_t q = (size_t)(i + u) * WPIX + j;
      for (int v = 0; v < 11; ++v)
        acc += (x0[q + v] + x1[q + v]) * w[u * 11 + v];
    }
  }
  float rr = acc + bias[0];
  out[tid]       = rr;
  out[OSZ + tid] = rr;
}

// ---------------------------------------------------------------------------
extern "C" void kernel_launch(void* const* d_in, const int* in_sizes, int n_in,
                              void* d_out, int out_size, void* d_ws,
                              size_t ws_size, hipStream_t stream) {
  const float* x    = (const float*)d_in[0];  // (2,256,512,512) fp32
  const float* kern = (const float*)d_in[1];  // (1,256,11,11)  fp32
  const float* bias = (const float*)d_in[2];  // (1,)           fp32
  float* out = (float*)d_out;                 // (2,1,502,502)  fp32

  const size_t need = PW_BYTES + (size_t)NPADT * HW * sizeof(_Float16);
  if (ws_size < need) {
    conv_naive<<<(OSZ + 255) / 256, 256, 0, stream>>>(x, kern, bias, out);
    return;
  }

  __bf16*    pw      = (__bf16*)d_ws;
  _Float16*  partial = (_Float16*)((char*)d_ws + PW_BYTES);

  prep_weights<<<8, 256, 0, stream>>>(kern, pw);
  conv_gemm<<<HW / QPB, 256, 0, stream>>>(x, pw, partial);
  conv_gather<<<(OSZ + 255) / 256, 256, 0, stream>>>(partial, bias, out);
}
